// StandardMRALSTM_37065567764562
// MI455X (gfx1250) — compile-verified, matrix operand order checked by reading, not run on hardware
//
#include <hip/hip_runtime.h>
#include <hip/hip_bf16.h>

typedef __attribute__((ext_vector_type(16))) __bf16 v16bf;
typedef __attribute__((ext_vector_type(8)))  __bf16 v8bf;
typedef __attribute__((ext_vector_type(8)))  float  v8f;

#define L_   4
#define H_   512
#define T_   128
#define G_   16
#define B_   64
#define F_   64
#define FH_  2560      // 5*H
#define NT_CELL 160    // 2560/16 output tiles (N)
#define KC_H 16        // 512/32 K-chunks
#define MT_B 4         // 64/16 output tiles (M)
#define PH_LAYER 32768 // MT_B*KC_H*32*16 packed bf16 per (slot,layer)
#define GRID_P  20     // persistent recurrence grid (20 blocks * 8 waves = 160 tiles)

__device__ __forceinline__ float sigmf(float x){ return 1.0f/(1.0f+__expf(-x)); }

__device__ __forceinline__ v8f wmma_bf16(v16bf a, v16bf b, v8f c){
  return __builtin_amdgcn_wmma_f32_16x16x32_bf16(false, a, false, b, (short)0, c, false, false);
}

// masks m1..m4 are one-hot scalars per (t,k), derived from the s chain
__device__ __forceinline__ void cell_masks(const float* ssum, const float* btf,
                                           int t, int k, int& m1,int& m2,int& m3,int& m4){
  const float inv = 1.0f/32768.0f;   // 1/(B*H)
  float lower = (k==0) ? btf[0*T_ + t]
                       : ((ssum[(k-1)*T_ + t]*inv > 0.5f) ? 1.f : 0.f);
  float prev  = (t==0) ? btf[k*T_ + 0]
                       : ((ssum[k*T_ + (t-1)]*inv > 0.5f) ? 1.f : 0.f);
  bool p = prev >= 0.5f, l = lower >= 0.5f;
  m1 = (!p &&  l); m2 = ( p && !l); m3 = ( p &&  l); m4 = (!p && !l);
}

// K offset inside a 32-wide chunk for (lane,e) in A/B fragment layout
__device__ __forceinline__ int koff(int lane, int e){
  return ((lane<16)?0:8) + ((e<8)? e : e+8);
}

// ticket-based device-wide barrier for the persistent recurrence kernel.
// GRID_P blocks are trivially co-resident on MI455X.
__device__ __forceinline__ void gbar(unsigned* bar){
  __syncthreads();
  __threadfence();                       // release: make phase writes visible
  if (threadIdx.x == 0){
    unsigned ticket = atomicAdd(bar, 1u);
    unsigned target = (ticket/GRID_P + 1u)*GRID_P;
    while (*(volatile unsigned*)bar < target){
      __builtin_amdgcn_s_sleep(1);
    }
  }
  __syncthreads();
  __threadfence();                       // acquire: invalidate stale L0 lines
}

// ---------------- prologue: zero state, bt, z-flags, pack xg fragments -------
__global__ void prep0(const float* x, const float* boundary,
                      float* ssum, float* btf, float* zf, float* cst,
                      __bf16* ph, __bf16* xa, unsigned* bar){
  const int N_BAR  = 4;
  const int N_SSUM = L_*T_;
  const int N_BT   = L_*T_;
  const int N_CST  = L_*B_*H_;
  const int N_PH   = 2*L_*PH_LAYER;
  const int N_Z    = T_*L_*B_;
  const int N_XA   = T_*L_*MT_B*32*16;
  const int total  = N_BAR+N_SSUM+N_BT+N_CST+N_PH+N_Z+N_XA;
  for (int i = blockIdx.x*blockDim.x + threadIdx.x; i < total; i += gridDim.x*blockDim.x){
    int r = i;
    if (r < N_BAR){ bar[r] = 0u; continue; } r -= N_BAR;
    if (r < N_SSUM){ ssum[r] = 0.f; continue; } r -= N_SSUM;
    if (r < N_BT){ btf[r] = (boundary[r] > 0.5f) ? 1.f : 0.f; continue; } r -= N_BT;
    if (r < N_CST){ cst[r] = 0.f; continue; } r -= N_CST;
    if (r < N_PH){ ph[r] = (__bf16)0.f; continue; } r -= N_PH;
    if (r < N_Z){
      int t = r/(L_*B_); int k = (r/B_)%L_; int b = r%B_;
      bool any = false;
      for (int f=0; f<G_; ++f){
        float v = x[((size_t)b*T_ + t)*F_ + k*G_ + f];
        any = any || !(v != v);
      }
      zf[r] = any ? 1.f : 0.f; continue;
    } r -= N_Z;
    { // xg A-fragments, K padded 16->32 with zeros
      int e = r & 15, lane = (r>>4)&31, mt = (r>>9)&3, k = (r>>11)&3, t = r>>13;
      int K = koff(lane, e); int m = mt*16 + (lane&15);
      float val = 0.f;
      if (K < G_){
        float v = x[((size_t)m*T_ + t)*F_ + k*G_ + K];
        val = (v != v) ? 0.f : v;
      }
      xa[r] = (__bf16)val; // layout ((t*L+k)*MT+mt)*512 + lane*16 + e == r
    }
  }
}

// ---------------- pack K=512 weights (W,Z,V,U[:, :512]) to bf16 B-frag layout
__global__ void pack_w512(const float* W, const float* Z, const float* V, const float* U,
                          __bf16* pw, __bf16* pz, __bf16* pv, __bf16* puh){
  const size_t PER = (size_t)L_*NT_CELL*KC_H*512; // 5,242,880
  const size_t total = 4*PER;
  for (size_t i = blockIdx.x*(size_t)blockDim.x + threadIdx.x; i < total;
       i += (size_t)gridDim.x*blockDim.x){
    int mat = (int)(i / PER); size_t rem = i % PER;
    int kL = (int)(rem / (NT_CELL*KC_H*512)); size_t r2 = rem % (NT_CELL*KC_H*512);
    int nt = (int)(r2 / (KC_H*512)); size_t r3 = r2 % (KC_H*512);
    int kc = (int)(r3 / 512); int r4 = (int)(r3 % 512);
    int lane = r4>>4, e = r4&15;
    int n = nt*16 + (lane&15);
    int K = kc*32 + koff(lane, e);
    float v;
    if      (mat==0) v = W[((size_t)kL*FH_ + n)*H_ + K];
    else if (mat==1) v = Z[((size_t)kL*FH_ + n)*H_ + K];
    else if (mat==2) v = V[((size_t)kL*FH_ + n)*H_ + K];
    else             v = U[((size_t)kL*FH_ + n)*(H_+G_) + K];
    __bf16* dst = (mat==0)?pw:(mat==1)?pz:(mat==2)?pv:puh;
    dst[rem] = (__bf16)v;
  }
}

// ---------------- pack K=16 weights (U[:,512:], J) padded to K=32, and Wo^T --
__global__ void pack_small(const float* U, const float* J, const float* Wo,
                           __bf16* pux, __bf16* pj, __bf16* pwo){
  const size_t PER = (size_t)L_*NT_CELL*512;  // 327,680
  const size_t N_WO = 4*16*512;               // 32,768
  const size_t total = 2*PER + N_WO;
  for (size_t i = blockIdx.x*(size_t)blockDim.x + threadIdx.x; i < total;
       i += (size_t)gridDim.x*blockDim.x){
    if (i < 2*PER){
      int mat = (int)(i / PER); size_t rem = i % PER;
      int kL = (int)(rem / (NT_CELL*512)); size_t r2 = rem % (NT_CELL*512);
      int nt = (int)(r2 / 512); int r4 = (int)(r2 % 512);
      int lane = r4>>4, e = r4&15;
      int n = nt*16 + (lane&15);
      int K = koff(lane, e);
      float v = 0.f;
      if (K < G_){
        v = (mat==0) ? U[((size_t)kL*FH_ + n)*(H_+G_) + H_ + K]
                     : J[((size_t)kL*FH_ + n)*G_ + K];
      }
      ((mat==0)?pux:pj)[rem] = (__bf16)v;
    } else {
      size_t rem = i - 2*PER;
      int nt = (int)(rem / (16*512)); size_t r3 = rem % (16*512);
      int kc = (int)(r3 / 512); int r4 = (int)(r3 % 512);
      int lane = r4>>4, e = r4&15;
      int n = nt*16 + (lane&15);
      int K = kc*32 + koff(lane, e);
      pwo[rem] = (__bf16)Wo[(size_t)n*H_ + K];   // B[K][n] = Wo[n][K]
    }
  }
}

// ---------------- pack Q (K=2048) and R (K=512) B-frags ----------------------
__global__ void pack_qr(const float* Q, const float* R, __bf16* pq, __bf16* pr){
  const size_t N_Q = (size_t)L_*32*64*512;  // 4,194,304
  const size_t N_R = (size_t)L_*32*16*512;  // 1,048,576
  const size_t total = N_Q + N_R;
  for (size_t i = blockIdx.x*(size_t)blockDim.x + threadIdx.x; i < total;
       i += (size_t)gridDim.x*blockDim.x){
    if (i < N_Q){
      size_t rem = i;
      int kL = (int)(rem / (32u*64u*512u)); size_t r2 = rem % (32u*64u*512u);
      int nt = (int)(r2 / (64*512)); size_t r3 = r2 % (64*512);
      int kc = (int)(r3 / 512); int r4 = (int)(r3 % 512);
      int lane = r4>>4, e = r4&15;
      int n = nt*16 + (lane&15);
      int K = kc*32 + koff(lane, e);
      pq[rem] = (__bf16)Q[((size_t)kL*2048 + K)*H_ + n];
    } else {
      size_t rem = i - N_Q;
      int kL = (int)(rem / (32u*16u*512u)); size_t r2 = rem % (32u*16u*512u);
      int nt = (int)(r2 / (16*512)); size_t r3 = r2 % (16*512);
      int kc = (int)(r3 / 512); int r4 = (int)(r3 % 512);
      int lane = r4>>4, e = r4&15;
      int n = nt*16 + (lane&15);
      int K = kc*32 + koff(lane, e);
      pr[rem] = (__bf16)R[((size_t)kL*H_ + K)*H_ + n];
    }
  }
}

// ---------------- persistent recurrence kernel -------------------------------
// One launch for all 128 timesteps x 4 layers. 20 blocks x 8 waves = 160
// waves = 160 N-tiles in the GEMM phase; device-wide ticket barrier between
// GEMM and gate phases replaces 1024 kernel launches.
__global__ void __launch_bounds__(256, 1)
cell_persist(__bf16* ph, const __bf16* xa,
             const __bf16* pw, const __bf16* pz, const __bf16* pv,
             const __bf16* puh, const __bf16* pux, const __bf16* pj,
             float* ssum, const float* btf, const float* zf,
             const float* bias, float* preb, float* cst, float* hh,
             __bf16* hcat, unsigned* bar){
  const int lane = threadIdx.x & 31;
  const int wave = threadIdx.x >> 5;
  const int n_tile = blockIdx.x*8 + wave;              // 0..159
  const int gtid = blockIdx.x*blockDim.x + threadIdx.x; // 0..5119

  for (int t = 0; t < T_; ++t){
    for (int k = 0; k < L_; ++k){
      int m1,m2,m3,m4; cell_masks(ssum, btf, t, k, m1,m2,m3,m4);
      const int kn = (k+1 < L_) ? k+1 : L_-1;
      const int slotP = (t+1) & 1, slotC = t & 1;

      // ---------------- GEMM phase (skipped entirely when m4) ----------------
      if (!m4){
        v8f accM[MT_B] = {}; v8f accU[MT_B] = {}; v8f accV[MT_B] = {};

        // main term: m1 -> h_prev[k] @ W^T ; m2|m3 -> h_prev[kn] @ Z^T
        {
          const __bf16* aB = ph + (size_t)(slotP*L_ + (m1 ? k : kn))*PH_LAYER;
          const __bf16* bB = (m1 ? pw : pz) + ((size_t)(k*NT_CELL + n_tile))*(KC_H*512);
          for (int kc=0; kc<KC_H; ++kc){
            if (kc+1 < KC_H) __builtin_prefetch(bB + (kc+1)*512 + lane*16, 0, 3);
            v16bf bf = *(const v16bf*)(bB + kc*512 + lane*16);
            #pragma unroll
            for (int mt=0; mt<MT_B; ++mt){
              v16bf af = *(const v16bf*)(aB + ((mt*KC_H + kc)*32 + lane)*16);
              accM[mt] = wmma_bf16(af, bf, accM[mt]);
            }
          }
        }

        if (m1 || m3){
          // t2 = z ? aU : aV ; aU = h_low @ Uh^T + xg @ Ux^T ; aV = h_low @ V^T
          if (k > 0){
            const __bf16* aB = ph + (size_t)(slotC*L_ + (k-1))*PH_LAYER;
            const __bf16* bU = puh + ((size_t)(k*NT_CELL + n_tile))*(KC_H*512);
            const __bf16* bV = pv  + ((size_t)(k*NT_CELL + n_tile))*(KC_H*512);
            for (int kc=0; kc<KC_H; ++kc){
              if (kc+1 < KC_H){
                __builtin_prefetch(bU + (kc+1)*512 + lane*16, 0, 3);
                __builtin_prefetch(bV + (kc+1)*512 + lane*16, 0, 3);
              }
              v16bf bu = *(const v16bf*)(bU + kc*512 + lane*16);
              v16bf bv = *(const v16bf*)(bV + kc*512 + lane*16);
              #pragma unroll
              for (int mt=0; mt<MT_B; ++mt){
                v16bf af = *(const v16bf*)(aB + ((mt*KC_H + kc)*32 + lane)*16);
                accU[mt] = wmma_bf16(af, bu, accU[mt]);
                accV[mt] = wmma_bf16(af, bv, accV[mt]);
              }
            }
          }
          v16bf bx = *(const v16bf*)(pux + ((size_t)(k*NT_CELL + n_tile))*512 + lane*16);
          const __bf16* xaB = xa + ((size_t)(t*L_ + k))*(MT_B*512);
          #pragma unroll
          for (int mt=0; mt<MT_B; ++mt){
            v16bf af = *(const v16bf*)(xaB + mt*512 + lane*16);
            accU[mt] = wmma_bf16(af, bx, accU[mt]);
          }
        } else { // m2: t2J = z ? aJ : 0  (accV stays 0)
          v16bf bj = *(const v16bf*)(pj + ((size_t)(k*NT_CELL + n_tile))*512 + lane*16);
          const __bf16* xaB = xa + ((size_t)(t*L_ + k))*(MT_B*512);
          #pragma unroll
          for (int mt=0; mt<MT_B; ++mt){
            v16bf af = *(const v16bf*)(xaB + mt*512 + lane*16);
            accU[mt] = wmma_bf16(af, bj, accU[mt]);
          }
        }

        // epilogue: pre = bias + main + (z ? U-path : V-path)
        const int n = n_tile*16 + (lane&15);
        const int rowoff = (lane>=16) ? 8 : 0;
        const float bn = bias[(size_t)k*FH_ + n];
        const float* zrow = zf + ((size_t)t*L_ + k)*B_;
        #pragma unroll
        for (int mt=0; mt<MT_B; ++mt){
          #pragma unroll
          for (int v=0; v<8; ++v){
            int m = mt*16 + rowoff + v;
            float sel = (zrow[m] > 0.5f) ? accU[mt][v] : accV[mt][v];
            preb[(size_t)m*FH_ + n] = bn + accM[mt][v] + sel;
          }
        }
      }

      gbar(bar);   // preb complete, visible device-wide

      // ---------------- gate phase ------------------------------------------
      {
        float local = 0.f;
        for (int idx = gtid; idx < B_*H_; idx += GRID_P*256){
          int b = idx >> 9, h = idx & (H_-1);
          float cp = cst[((size_t)k*B_ + b)*H_ + h];
          float hp = (t>0) ? hh[(((size_t)k*B_ + b)*T_ + (t-1))*H_ + h] : 0.f;

          float hN, cN, sgpre;
          if (m4){
            cN = cp;
            float o = sigmf(bias[(size_t)k*FH_ + 3*H_ + h]);
            hN = o * tanhf(cN);
            sgpre = bias[(size_t)k*FH_ + 4*H_ + h];
          } else {
            float pf = m1 ? preb[(size_t)b*FH_ + h] : bias[(size_t)k*FH_ + h];
            float f = sigmf(pf);
            float g = tanhf(preb[(size_t)b*FH_ + H_   + h]);
            float i = sigmf(preb[(size_t)b*FH_ + 2*H_ + h]);
            cN = m1 ? (f*cp + i*g) : (i*g);
            hN = hp;                               // h only updates in m4 branch
            sgpre = preb[(size_t)b*FH_ + 4*H_ + h];
          }
          float sg = (sgpre + 1.f)*0.5f;
          local += fminf(fmaxf(sg, 0.f), 1.f);

          cst[((size_t)k*B_ + b)*H_ + h] = cN;
          hh[(((size_t)k*B_ + b)*T_ + t)*H_ + h] = hN;
          hcat[((size_t)b*T_ + t)*(L_*H_) + k*H_ + h] = (__bf16)hN;

          // re-pack h into WMMA A-fragment layout (t-parity double buffer)
          int mt = b>>4, kc = h>>5, r = h&31;
          int hi = (r & 8) ? 1 : 0;
          int plane = (b&15) + (hi ? 16 : 0);
          int e = (r&7) + ((r>=16) ? 8 : 0);
          ph[(size_t)((t&1)*L_ + k)*PH_LAYER + (mt*KC_H + kc)*512 + plane*16 + e] = (__bf16)hN;
        }
        #pragma unroll
        for (int off=16; off; off>>=1) local += __shfl_down(local, off, 32);
        if (lane == 0) atomicAdd(&ssum[k*T_ + t], local);
      }

      gbar(bar);   // state + s-sum visible before next (t,k)
    }
  }
}

// ---------------- excitation: r = sigmoid(hcat @ Q[k]) -----------------------
__global__ void exc_r(const __bf16* hcat, const __bf16* pq, __bf16* rbuf, int k){
  const int lane = threadIdx.x & 31;
  int job = blockIdx.x*8 + (threadIdx.x>>5);   // 16384 = 512 m-tiles * 32 n-tiles
  int m_tile = job >> 5, n_tile = job & 31;
  int m = m_tile*16 + (lane&15);
  v8f acc = {};
  const __bf16* bB = pq + ((size_t)(k*32 + n_tile))*(64*512);
  for (int kc=0; kc<64; ++kc){
    if (kc+1 < 64) __builtin_prefetch(bB + (kc+1)*512 + lane*16, 0, 3);
    int kof = kc*32 + ((lane<16)?0:8);
    v8bf lo = *(const v8bf*)(hcat + (size_t)m*2048 + kof);
    v8bf hi = *(const v8bf*)(hcat + (size_t)m*2048 + kof + 16);
    v16bf a = __builtin_shufflevector(lo, hi, 0,1,2,3,4,5,6,7,8,9,10,11,12,13,14,15);
    v16bf bf = *(const v16bf*)(bB + kc*512 + lane*16);
    acc = wmma_bf16(a, bf, acc);
  }
  int col = n_tile*16 + (lane&15);
  int rowb = m_tile*16 + ((lane>=16)?8:0);
  #pragma unroll
  for (int v=0; v<8; ++v)
    rbuf[(size_t)(rowb+v)*H_ + col] = (__bf16)sigmf(acc[v]);
}

// ---------------- excitation: acc += (r @ R[k]) * h_bg[k] --------------------
__global__ void exc_mul(const __bf16* rbuf, const __bf16* pr, const float* hh,
                        float* accb, int k){
  const int lane = threadIdx.x & 31;
  int job = blockIdx.x*8 + (threadIdx.x>>5);   // 16384 tiles
  int m_tile = job >> 5, n_tile = job & 31;
  int m = m_tile*16 + (lane&15);
  v8f acc = {};
  const __bf16* bB = pr + ((size_t)(k*32 + n_tile))*(16*512);
  for (int kc=0; kc<16; ++kc){
    int kof = kc*32 + ((lane<16)?0:8);
    v8bf lo = *(const v8bf*)(rbuf + (size_t)m*H_ + kof);
    v8bf hi = *(const v8bf*)(rbuf + (size_t)m*H_ + kof + 16);
    v16bf a = __builtin_shufflevector(lo, hi, 0,1,2,3,4,5,6,7,8,9,10,11,12,13,14,15);
    v16bf bf = *(const v16bf*)(bB + kc*512 + lane*16);
    acc = wmma_bf16(a, bf, acc);
  }
  int col = n_tile*16 + (lane&15);
  int rowb = m_tile*16 + ((lane>=16)?8:0);
  #pragma unroll
  for (int v=0; v<8; ++v){
    int row = rowb + v;                        // row = b*T + t
    int b = row >> 7, t = row & (T_-1);
    float val = acc[v] * hh[(((size_t)k*B_ + b)*T_ + t)*H_ + col];
    if (k == 0) accb[(size_t)row*H_ + col]  = val;
    else        accb[(size_t)row*H_ + col] += val;
  }
}

// ---------------- final: out = relu(acc) @ Wo^T + bo -------------------------
__global__ void exc_out(const float* accb, const __bf16* pwo, const float* bo,
                        float* out){
  const int lane = threadIdx.x & 31;
  int job = blockIdx.x*8 + (threadIdx.x>>5);   // 2048 = 512 m-tiles * 4 n-tiles
  int m_tile = job >> 2, n_tile = job & 3;
  int m = m_tile*16 + (lane&15);
  v8f acc = {};
  for (int kc=0; kc<16; ++kc){
    int kof = kc*32 + ((lane<16)?0:8);
    v8f lo = *(const v8f*)(accb + (size_t)m*H_ + kof);
    v8f hi = *(const v8f*)(accb + (size_t)m*H_ + kof + 16);
    v16bf a;
    #pragma unroll
    for (int i=0;i<8;++i){
      a[i]   = (__bf16)fmaxf(lo[i], 0.f);
      a[i+8] = (__bf16)fmaxf(hi[i], 0.f);
    }
    v16bf bf = *(const v16bf*)(pwo + (size_t)(n_tile*16 + kc)*512 + lane*16);
    acc = wmma_bf16(a, bf, acc);
  }
  int col = n_tile*16 + (lane&15);
  int rowb = m_tile*16 + ((lane>=16)?8:0);
  #pragma unroll
  for (int v=0; v<8; ++v)
    out[(size_t)(rowb+v)*F_ + col] = acc[v] + bo[col];
}

// ============================== host side ====================================
extern "C" void kernel_launch(void* const* d_in, const int* in_sizes, int n_in,
                              void* d_out, int out_size, void* d_ws, size_t ws_size,
                              hipStream_t stream){
  (void)in_sizes; (void)n_in; (void)out_size; (void)ws_size;
  const float* x        = (const float*)d_in[0];
  const float* W        = (const float*)d_in[1];
  const float* Z        = (const float*)d_in[2];
  const float* U        = (const float*)d_in[3];
  const float* V        = (const float*)d_in[4];
  const float* J        = (const float*)d_in[5];
  const float* bias     = (const float*)d_in[6];
  const float* boundary = (const float*)d_in[7];
  const float* Q        = (const float*)d_in[8];
  const float* R        = (const float*)d_in[9];
  const float* Wo       = (const float*)d_in[10];
  const float* bo       = (const float*)d_in[11];
  float* out = (float*)d_out;

  char* ws = (char*)d_ws;
  size_t off = 0;
  auto alloc = [&](size_t bytes)->char*{
    char* p = ws + off; off = (off + bytes + 255) & ~(size_t)255; return p;
  };
  unsigned* bar  = (unsigned*)alloc(4*4);
  float*  ssum = (float*) alloc(L_*T_*4);
  float*  btf  = (float*) alloc(L_*T_*4);
  float*  zf   = (float*) alloc((size_t)T_*L_*B_*4);
  float*  cst  = (float*) alloc((size_t)L_*B_*H_*4);
  __bf16* ph   = (__bf16*)alloc((size_t)2*L_*PH_LAYER*2);
  __bf16* xa   = (__bf16*)alloc((size_t)T_*L_*MT_B*512*2);
  float*  preb = (float*) alloc((size_t)B_*FH_*4);
  __bf16* pw   = (__bf16*)alloc((size_t)L_*NT_CELL*KC_H*512*2);
  __bf16* pz   = (__bf16*)alloc((size_t)L_*NT_CELL*KC_H*512*2);
  __bf16* pv   = (__bf16*)alloc((size_t)L_*NT_CELL*KC_H*512*2);
  __bf16* puh  = (__bf16*)alloc((size_t)L_*NT_CELL*KC_H*512*2);
  __bf16* pux  = (__bf16*)alloc((size_t)L_*NT_CELL*512*2);
  __bf16* pj   = (__bf16*)alloc((size_t)L_*NT_CELL*512*2);
  __bf16* pwo  = (__bf16*)alloc((size_t)4*16*512*2);
  __bf16* pq   = (__bf16*)alloc((size_t)L_*32*64*512*2);
  __bf16* pr   = (__bf16*)alloc((size_t)L_*32*16*512*2);
  float*  hh   = (float*) alloc((size_t)L_*B_*T_*H_*4);
  __bf16* hcat = (__bf16*)alloc((size_t)B_*T_*L_*H_*2);
  __bf16* rbuf = (__bf16*)alloc((size_t)B_*T_*H_*2);
  float*  accb = (float*) alloc((size_t)B_*T_*H_*4);

  // prologue: state init + bf16 weight pre-pack (once per call; weights then
  // stay L2-resident for the 128-step recurrence)
  prep0     <<<2048, 256, 0, stream>>>(x, boundary, ssum, btf, zf, cst, ph, xa, bar);
  pack_w512 <<<8192, 256, 0, stream>>>(W, Z, V, U, pw, pz, pv, puh);
  pack_small<<< 512, 256, 0, stream>>>(U, J, Wo, pux, pj, pwo);
  pack_qr   <<<4096, 256, 0, stream>>>(Q, R, pq, pr);

  // entire recurrence: ONE persistent kernel (device-wide barriers between
  // phases replace 1024 per-(t,k) kernel launches)
  cell_persist<<<GRID_P, 256, 0, stream>>>(ph, xa, pw, pz, pv, puh, pux, pj,
                                           ssum, btf, zf, bias, preb, cst, hh,
                                           hcat, bar);

  // excitation block (wide, parallel WMMA launches)
  for (int k = 0; k < L_; ++k){
    exc_r  <<<2048, 256, 0, stream>>>(hcat, pq, rbuf, k);
    exc_mul<<<2048, 256, 0, stream>>>(rbuf, pr, hh, accb, k);
  }
  exc_out<<<256, 256, 0, stream>>>(accb, pwo, bo, out);
}
